// TopKRouter_87368224735636
// MI455X (gfx1250) — matrix operand order repaired and checked by reading, not compiled, not run on hardware
//
#include <hip/hip_runtime.h>
#include <hip/hip_bf16.h>

// ---------------- problem constants (from reference) ----------------
#define DD          512                // model dim
#define SLEN        4096               // sequence length per batch
#define KR          8                  // top-k routes
#define TILE        16                 // WMMA M/N tile
#define GROUP_ROWS  128                // k rows staged per TDM op (8 waves x 16)
#define NIT         (SLEN / GROUP_ROWS)// 32 iterations

typedef __attribute__((ext_vector_type(16))) __bf16 v16bf;
typedef __attribute__((ext_vector_type(8)))  float  v8f;
typedef __attribute__((ext_vector_type(4)))  __bf16 v4bf;
typedef __attribute__((ext_vector_type(4)))  float  f32x4;
typedef __attribute__((ext_vector_type(4)))  unsigned int u32x4;
typedef __attribute__((ext_vector_type(4)))  int    i32x4;
typedef __attribute__((ext_vector_type(8)))  int    i32x8;

// ---- A-matrix fragment (16x32 bf16), ISA 7.12.2 16-bit A layout ----
__device__ __forceinline__ v16bf load_fragA(const __bf16* base, int row0,
                                            int ld, int k0, int lane) {
  int l = lane & 15, hi = lane >> 4;
  const __bf16* p = base + (size_t)(row0 + l) * ld + k0 + hi * 8;
  union { u32x4 u[2]; v16bf v; } U;
  U.u[0] = *reinterpret_cast<const u32x4*>(p);
  U.u[1] = *reinterpret_cast<const u32x4*>(p + 16);
  return U.v;
}

// ---- B-matrix fragment (32x16 bf16): col=lane&15 owns one source row,
// lane<16 holds K=k0..k0+7,k0+16..k0+23; lane>=16 holds K=k0+8..k0+15,k0+24..k0+31 ----
__device__ __forceinline__ v16bf load_fragB(const __bf16* base, int col0,
                                            int ld, int k0, int lane) {
  int l = lane & 15, hi = lane >> 4;
  const __bf16* p = base + (size_t)(col0 + l) * ld + k0 + hi * 8;
  union { u32x4 u[2]; v16bf v; } U;
  U.u[0] = *reinterpret_cast<const u32x4*>(p);
  U.u[1] = *reinterpret_cast<const u32x4*>(p + 16);
  return U.v;
}

// low 32 bits of a generic LDS pointer == wave-relative LDS byte offset
__device__ __forceinline__ unsigned lds_offset(const void* p) {
  return (unsigned)(uintptr_t)p;
}

// ---- TDM: DMA GROUP_ROWS x 512 bf16 (row-major, stride 512) into LDS ----
// D# per ISA cdna5 ch.8: group0 = {count, lds_addr, global_addr, type=2},
// group1 = {data_size=2B, tensor_dim0/1, tile_dim0/1, dim0_stride}.
__device__ __forceinline__ void tdm_load_rows_to_lds(unsigned lds_off,
                                                     const __bf16* gptr) {
  unsigned long long ga = (unsigned long long)(uintptr_t)gptr;
  u32x4 g0;
  g0[0] = 1u;                                            // count=1 (valid D#)
  g0[1] = lds_off;                                       // lds_addr [63:32]
  g0[2] = (unsigned)ga;                                  // global_addr[31:0]
  g0[3] = (unsigned)((ga >> 32) & 0x01FFFFFFull) | 0x80000000u; // [56:32]|type=2
  i32x8 g1;
  g1[0] = 0x00010000;                 // workgroup_mask=0, data_size=1 (2 bytes)
  g1[1] = (int)((unsigned)DD << 16);  // tensor_dim0 = 512 (bits 79:48)
  g1[2] = (int)((unsigned)SLEN << 16);// tensor_dim1 = 4096 (bits 111:80)
  g1[3] = (int)((unsigned)DD << 16);  // tile_dim0 = 512 (bits 127:112)
  g1[4] = GROUP_ROWS;                 // tile_dim1 = 128 (bits 143:128)
  g1[5] = DD;                         // tensor_dim0_stride = 512 (bits 207:160)
  g1[6] = 0;
  g1[7] = 0;
  i32x4 z4 = {0, 0, 0, 0};
#if __clang_major__ >= 23
  i32x8 z8 = {0, 0, 0, 0, 0, 0, 0, 0};
  __builtin_amdgcn_tensor_load_to_lds(g0, g1, z4, z4, z8, 0);
#else
  __builtin_amdgcn_tensor_load_to_lds(g0, g1, z4, z4, 0);
#endif
}

// ---------------- f32 -> bf16 conversion (4-wide) ----------------
__global__ void cvt4_bf16_kernel(const float* __restrict__ in,
                                 __bf16* __restrict__ out, int n4) {
  int i = blockIdx.x * blockDim.x + threadIdx.x;
  if (i < n4) {
    f32x4 x = *reinterpret_cast<const f32x4*>(in + 4 * (size_t)i);
    v4bf o;
    o[0] = (__bf16)x[0]; o[1] = (__bf16)x[1];
    o[2] = (__bf16)x[2]; o[3] = (__bf16)x[3];
    *reinterpret_cast<v4bf*>(out + 4 * (size_t)i) = o;
  }
}

// ---------------- q_proj = q @ w_score^T + b_score  (bf16 out) ----------------
__global__ __launch_bounds__(256) void qproj_kernel(
    const float*  __restrict__ q,       // [BS, D] f32
    const __bf16* __restrict__ wsc,     // [D, D]  bf16 (row = output col)
    const float*  __restrict__ b_score, // [D]
    __bf16*       __restrict__ qp) {    // [BS, D] bf16
  __shared__ __bf16 qs[TILE * DD];      // 16 KB
  const int tid  = threadIdx.x;
  const int row0 = blockIdx.x * TILE;

  for (int i = tid; i < TILE * DD; i += 256)
    qs[i] = (__bf16)q[(size_t)row0 * DD + i];
  __syncthreads();

  const int wave = tid >> 5, lane = tid & 31;
  const int l = lane & 15, hi = lane >> 4;

  for (int ct = 0; ct < 4; ++ct) {
    const int col0 = (wave * 4 + ct) * TILE;
    v8f c = {};
    for (int kk = 0; kk < DD; kk += 32) {
      v16bf a = load_fragA(qs,  0,    DD, kk, lane);
      v16bf b = load_fragB(wsc, col0, DD, kk, lane);
      c = __builtin_amdgcn_wmma_f32_16x16x32_bf16(false, a, false, b,
                                                  (short)0, c, false, false);
    }
    const int col = col0 + l;
    const float bias = b_score[col];
#pragma unroll
    for (int r = 0; r < 8; ++r) {
      const int row = row0 + r + 8 * hi;
      qp[(size_t)row * DD + col] = (__bf16)(c[r] + bias);
    }
  }
}

// -------- fused scores + top-8 + softmax + gather, TDM-staged k --------
// block = (batch, 16-row tile); A tile lives in registers (16 frags);
// wave0 TDM-DMAs 128 k rows/iteration into a 2x128KB LDS double buffer.
__global__ __launch_bounds__(256, 1) void route_kernel(
    const __bf16* __restrict__ qp,      // [BS, D] bf16 (projected q)
    const __bf16* __restrict__ kb,      // [BS, D] bf16
    const float*  __restrict__ v,       // [BS, D] f32
    int*          __restrict__ routes,  // [BS, 8]
    float*        __restrict__ weights, // [BS, 8]
    float*        __restrict__ outp) {  // [BS, D]
  extern __shared__ __bf16 kbuf[];        // 2 x GROUP_ROWS x DD bf16 = 256 KB
  __shared__ float  sc[8][TILE][TILE];    // 8 KB
  __shared__ float  cand_s[8][TILE][KR];  // 4 KB
  __shared__ int    cand_i[8][TILE][KR];  // 4 KB
  __shared__ float  wgt[TILE][KR];
  __shared__ int    rts[TILE][KR];

  const int tid  = threadIdx.x;
  const int wave = tid >> 5, lane = tid & 31;
  const int l = lane & 15, hi = lane >> 4;
  const int b  = blockIdx.x / (SLEN / TILE);
  const int rt = blockIdx.x % (SLEN / TILE);
  const size_t rowBase = (size_t)b * SLEN + (size_t)rt * TILE;
  const __bf16* kbatch = kb + (size_t)b * SLEN * DD;

  // A tile (16 x 512) held entirely in registers: 16 fragments, reused 256x.
  v16bf afrag[DD / 32];
#pragma unroll
  for (int kk = 0; kk < DD / 32; ++kk)
    afrag[kk] = load_fragA(qp + rowBase * DD, 0, DD, kk * 32, lane);

  if (wave == 0)  // prologue DMA: first 128 k rows -> buffer 0
    tdm_load_rows_to_lds(lds_offset(kbuf), kbatch);

  float ts[KR]; int ti[KR];
#pragma unroll
  for (int j = 0; j < KR; ++j) { ts[j] = -3.4e38f; ti[j] = 0; }
  const float scale = 0.044194173824159216f; // 1/sqrt(512)

  for (int it = 0; it < NIT; ++it) {
    if (wave == 0) {
      if (it + 1 < NIT) {  // keep next DMA in flight, wait only for current
        tdm_load_rows_to_lds(
            lds_offset(kbuf + (size_t)((it + 1) & 1) * GROUP_ROWS * DD),
            kbatch + (size_t)(it + 1) * GROUP_ROWS * DD);
        __builtin_amdgcn_s_wait_tensorcnt((short)1);
      } else {
        __builtin_amdgcn_s_wait_tensorcnt((short)0);
      }
    }
    __syncthreads();  // staged k rows visible to all waves

    const __bf16* bbuf = kbuf + (size_t)(it & 1) * GROUP_ROWS * DD;
    v8f c = {};
#pragma unroll
    for (int kk = 0; kk < DD / 32; ++kk) {
      v16bf bm = load_fragB(bbuf, wave * TILE, DD, kk * 32, lane);
      c = __builtin_amdgcn_wmma_f32_16x16x32_bf16(false, afrag[kk], false, bm,
                                                  (short)0, c, false, false);
    }
    const int t0 = it * GROUP_ROWS + wave * TILE;
#pragma unroll
    for (int r = 0; r < 8; ++r) sc[wave][r + 8 * hi][l] = c[r] * scale;

    if (lane < 16) {  // lane owns one row: insert 16 candidates
      for (int cidx = 0; cidx < TILE; ++cidx) {
        float s = sc[wave][lane][cidx];
        int  idx = t0 + cidx;
#pragma unroll
        for (int j = 0; j < KR; ++j) {
          if (s > ts[j]) {
            float tf = ts[j]; ts[j] = s; s = tf;
            int   tn = ti[j]; ti[j] = idx; idx = tn;
          }
        }
      }
    }
    __syncthreads();  // all waves done with bbuf before it is re-filled
  }

  if (lane < 16) {
#pragma unroll
    for (int j = 0; j < KR; ++j) {
      cand_s[wave][lane][j] = ts[j];
      cand_i[wave][lane][j] = ti[j];
    }
  }
  __syncthreads();

  if (tid < 16) {  // merge 8 wave-lists per row, softmax, emit routes/weights
    const int row = tid;
    float fs[KR]; int fi[KR];
#pragma unroll
    for (int j = 0; j < KR; ++j) { fs[j] = cand_s[0][row][j]; fi[j] = cand_i[0][row][j]; }
    for (int w = 1; w < 8; ++w) {
#pragma unroll
      for (int j2 = 0; j2 < KR; ++j2) {
        float s = cand_s[w][row][j2];
        int  idx = cand_i[w][row][j2];
#pragma unroll
        for (int j = 0; j < KR; ++j) {
          if (s > fs[j]) {
            float tf = fs[j]; fs[j] = s; s = tf;
            int   tn = fi[j]; fi[j] = idx; idx = tn;
          }
        }
      }
    }
    const float m = fs[0];  // sorted descending
    float e[KR]; float sum = 0.f;
#pragma unroll
    for (int j = 0; j < KR; ++j) { e[j] = __expf(fs[j] - m); sum += e[j]; }
    const float inv = 1.f / sum;
    const size_t o = (rowBase + row) * KR;
#pragma unroll
    for (int j = 0; j < KR; ++j) {
      const float wj = e[j] * inv;
      wgt[row][j] = wj;  rts[row][j] = fi[j];
      weights[o + j] = wj;
      routes[o + j]  = fi[j];
    }
  }
  __syncthreads();

  // output[row, :] = sum_j w_j * v[b, route_j, :]; 16 threads per row
  const int row = tid >> 4, c0 = tid & 15;
  float w8[KR]; int i8[KR];
#pragma unroll
  for (int j = 0; j < KR; ++j) { w8[j] = wgt[row][j]; i8[j] = rts[row][j]; }
  const float* vb = v + (size_t)b * SLEN * DD;
  float* orow = outp + (rowBase + row) * DD;
  for (int cc = 0; cc < DD / TILE; ++cc) {
    const int col = c0 + cc * TILE;
    float acc = 0.f;
#pragma unroll
    for (int j = 0; j < KR; ++j)
      acc += w8[j] * vb[(size_t)i8[j] * DD + col];
    orow[col] = acc;
  }
}

// ---------------- host launcher ----------------
extern "C" void kernel_launch(void* const* d_in, const int* in_sizes, int n_in,
                              void* d_out, int out_size, void* d_ws, size_t ws_size,
                              hipStream_t stream) {
  const float* q       = (const float*)d_in[0];
  const float* k       = (const float*)d_in[1];
  const float* v       = (const float*)d_in[2];
  // d_in[3] fingerprint, d_in[6] w_fp, d_in[7] b_fp: provably dead
  // (per-row-constant score shift; top-k order & softmax are shift-invariant).
  const float* w_score = (const float*)d_in[4];
  const float* b_score = (const float*)d_in[5];

  const int BS = in_sizes[0] / DD;  // B * S (= 16384)
  const int nK = BS * DD;

  char* wsb = (char*)d_ws;
  __bf16* kbf   = (__bf16*)wsb;                     // BS*D bf16
  __bf16* qpbf  = (__bf16*)(wsb + (size_t)nK * 2);  // BS*D bf16
  __bf16* wscbf = (__bf16*)(wsb + (size_t)nK * 4);  // D*D  bf16

  int*   routes  = (int*)d_out;
  float* weights = (float*)d_out + (size_t)BS * KR;
  float* outp    = (float*)d_out + (size_t)2 * BS * KR;

  static_cast<void>(hipFuncSetAttribute(
      reinterpret_cast<const void*>(route_kernel),
      hipFuncAttributeMaxDynamicSharedMemorySize, 2 * GROUP_ROWS * DD * 2));

  cvt4_bf16_kernel<<<(nK / 4 + 255) / 256, 256, 0, stream>>>(k, kbf, nK / 4);
  cvt4_bf16_kernel<<<(DD * DD / 4 + 255) / 256, 256, 0, stream>>>(
      w_score, wscbf, DD * DD / 4);
  qproj_kernel<<<BS / TILE, 256, 0, stream>>>(q, wscbf, b_score, qpbf);
  route_kernel<<<BS / TILE, 256, 2 * GROUP_ROWS * DD * 2, stream>>>(
      qpbf, kbf, v, routes, weights, outp);
}
